// SwinTransformerBlock_65180423685303
// MI455X (gfx1250) — matrix operand order repaired and checked by reading, not compiled, not run on hardware
//
#include <hip/hip_runtime.h>
#include <hip/hip_bf16.h>

// ---------------------------------------------------------------------------
// Swin Transformer block for MI455X (gfx1250, wave32, WMMA).
// GEMMs on v_wmma_f32_16x16x32_f16 (f16 operands, f32 accum). Contiguous
// global->LDS operand staging uses CDNA5 async-to-LDS (ASYNCcnt) instead of
// VGPR staging; transposed weight tiles go through VGPRs. LN / shift / mask /
// softmax / GELU / residual fused into GEMM prologues/epilogues.
// ---------------------------------------------------------------------------

typedef _Float16 v16h __attribute__((ext_vector_type(16)));
typedef _Float16 v8h  __attribute__((ext_vector_type(8)));
typedef float    v8f  __attribute__((ext_vector_type(8)));

union Frag16 { v16h f; v8h h[2]; };

#define BATCH   64
#define HH      56
#define CC      256
#define NHEAD   8
#define DHEAD   32
#define WSZ     7
#define SSZ     3
#define NTOK    49
#define MTOT    200704        // 64*64*49 window-tokens (== 64*3136)
#define QKV_STRIDE ((size_t)4096 * NHEAD * NTOK * DHEAD)

static __device__ __forceinline__ v8f wmma_f16(const Frag16& a, const Frag16& b, v8f c) {
  return __builtin_amdgcn_wmma_f32_16x16x32_f16(false, a.f, false, b.f,
                                                (short)0, c, false, false);
}

// CDNA5 async global->LDS copy of 16B (applies INST_OFFSET to both sides).
static __device__ __forceinline__ void async_b128(unsigned lds_byte_addr,
                                                  const void* gaddr) {
  asm volatile("global_load_async_to_lds_b128 %0, %1, off"
               :: "v"(lds_byte_addr), "v"(gaddr) : "memory");
}
static __device__ __forceinline__ void async_b128_o16(unsigned lds_byte_addr,
                                                      const void* gaddr) {
  asm volatile("global_load_async_to_lds_b128 %0, %1, off offset:16"
               :: "v"(lds_byte_addr), "v"(gaddr) : "memory");
}
static __device__ __forceinline__ void wait_async0() {
  asm volatile("s_wait_asynccnt 0" ::: "memory");
}
static __device__ __forceinline__ unsigned lds_addr32(const void* p) {
  return (unsigned)(unsigned long long)p;   // low 32 bits of generic LDS ptr
}

// ---------------------------------------------------------------------------
// f32 -> f16 copy (weights)
// ---------------------------------------------------------------------------
__global__ __launch_bounds__(256) void f32_to_f16_kernel(const float* __restrict__ src,
                                                         _Float16* __restrict__ dst, int n) {
  int i = blockIdx.x * 256 + threadIdx.x;
  if (i < n) dst[i] = (_Float16)src[i];
}

// ---------------------------------------------------------------------------
// LayerNorm (+ optional cyclic shift & window partition) -> f16 rows
// ---------------------------------------------------------------------------
template<bool WINDOWED>
__global__ __launch_bounds__(256) void ln_kernel(const float* __restrict__ x,
                                                 const float* __restrict__ g,
                                                 const float* __restrict__ b,
                                                 _Float16* __restrict__ out) {
  int row = blockIdx.x;
  size_t src;
  if (WINDOWED) {
    int w = row / NTOK, t = row % NTOK;
    int bb = w >> 6, wi = w & 63;
    int wy = wi >> 3, wx = wi & 7, iy = t / WSZ, ix = t % WSZ;
    int y  = wy * WSZ + iy + SSZ; if (y  >= HH) y  -= HH;   // roll(-3)
    int xp = wx * WSZ + ix + SSZ; if (xp >= HH) xp -= HH;
    src = ((size_t)bb * (HH * HH) + (size_t)y * HH + xp) * CC;
  } else {
    src = (size_t)row * CC;
  }
  int c = threadIdx.x;
  float v = x[src + c];
  __shared__ float red[256];
  red[c] = v; __syncthreads();
  for (int off = 128; off; off >>= 1) { if (c < off) red[c] += red[c + off]; __syncthreads(); }
  float mu = red[0] * (1.0f / CC);
  __syncthreads();
  float dv = v - mu;
  red[c] = dv * dv; __syncthreads();
  for (int off = 128; off; off >>= 1) { if (c < off) red[c] += red[c + off]; __syncthreads(); }
  float var = red[0] * (1.0f / CC);
  float o = dv * rsqrtf(var + 1e-5f) * g[c] + b[c];
  out[(size_t)row * CC + c] = (_Float16)o;
}

// ---------------------------------------------------------------------------
// Generic WMMA GEMM: D = A(f16, MxK) @ B(f16, KxN) + epilogue.
// Tile: 128x64, 256 threads = 8 waves, wave grid 4(M) x 2(N), 32x32 per wave.
// A tile staged via async global->LDS; B tile transposed through VGPRs.
// Epilogues: 0=QKV  1=PROJ  2=FC1(GELU)  3=FC2(residual add)
// ---------------------------------------------------------------------------
template<int EPI, int KTOT, int NTOT>
__global__ __launch_bounds__(256) void gemm_wmma_kernel(
    const _Float16* __restrict__ A, const _Float16* __restrict__ Bw,
    const float* __restrict__ bias,
    float* __restrict__ outF, _Float16* __restrict__ outH,
    const float* __restrict__ resid) {
  __shared__ __attribute__((aligned(16))) _Float16 lA[128 * 32];
  __shared__ __attribute__((aligned(16))) _Float16 lBt[64 * 32];   // transposed: [n][k]

  const int tid = threadIdx.x;
  const int m0 = blockIdx.y * 128;
  const int n0 = blockIdx.x * 64;
  const int wave = tid >> 5, lane = tid & 31;
  const int wm = (wave & 3) * 32, wn = (wave >> 2) * 32;
  const int lrow = lane & 15, lhi = lane >> 4;

  const int arow = tid >> 1, akh = (tid & 1) * 16;             // A-tile slot
  const unsigned aLds = lds_addr32(lA + arow * 32 + akh);      // 32B per thread
  const int bk = tid >> 3, bns = (tid & 7) * 8;                // B-tile slot

  v8f acc[2][2] = {};

  for (int k0 = 0; k0 < KTOT; k0 += 32) {
    // A tile: 128x32 halves, async DMA straight into LDS (2x b128 / thread).
    {
      const _Float16* src = A + (size_t)(m0 + arow) * KTOT + k0 + akh;
      async_b128(aLds, src);
      async_b128_o16(aLds, src);
    }
    // B tile: 32x64 -> store transposed (K-sequential per output column).
    {
      const _Float16* src = Bw + (size_t)(k0 + bk) * NTOT + n0 + bns;
#pragma unroll
      for (int i = 0; i < 8; ++i) lBt[(bns + i) * 32 + bk] = src[i];
      if (k0 + 32 < KTOT)
        __builtin_prefetch(Bw + (size_t)(k0 + 32 + bk) * NTOT + n0 + bns, 0, 1);
    }
    wait_async0();
    __syncthreads();

    Frag16 a0, a1, b0, b1;
    a0.h[0] = *(const v8h*)(lA + (wm + lrow) * 32 + lhi * 8);
    a0.h[1] = *(const v8h*)(lA + (wm + lrow) * 32 + 16 + lhi * 8);
    a1.h[0] = *(const v8h*)(lA + (wm + 16 + lrow) * 32 + lhi * 8);
    a1.h[1] = *(const v8h*)(lA + (wm + 16 + lrow) * 32 + 16 + lhi * 8);
    b0.h[0] = *(const v8h*)(lBt + (wn + lrow) * 32 + lhi * 8);
    b0.h[1] = *(const v8h*)(lBt + (wn + lrow) * 32 + 16 + lhi * 8);
    b1.h[0] = *(const v8h*)(lBt + (wn + 16 + lrow) * 32 + lhi * 8);
    b1.h[1] = *(const v8h*)(lBt + (wn + 16 + lrow) * 32 + 16 + lhi * 8);

    acc[0][0] = wmma_f16(a0, b0, acc[0][0]);
    acc[0][1] = wmma_f16(a0, b1, acc[0][1]);
    acc[1][0] = wmma_f16(a1, b0, acc[1][0]);
    acc[1][1] = wmma_f16(a1, b1, acc[1][1]);
    __syncthreads();
  }

  // Epilogue: acc element r maps to (row = r + lhi*8, col = lrow) in 16x16 tile.
  const float qscale = 0.17677669529663687f;  // 1/sqrt(32)
#pragma unroll
  for (int ti = 0; ti < 2; ++ti)
#pragma unroll
    for (int tj = 0; tj < 2; ++tj)
#pragma unroll
      for (int r = 0; r < 8; ++r) {
        int m = m0 + wm + ti * 16 + r + lhi * 8;
        int n = n0 + wn + tj * 16 + lrow;
        float val = acc[ti][tj][r] + bias[n];
        if constexpr (EPI == 0) {             // QKV scatter
          int s = n >> 8, hh = (n >> 5) & 7, dd = n & 31;
          if (s == 0) val *= qscale;
          int w = m / NTOK, t = m % NTOK;
          size_t idx = (size_t)s * QKV_STRIDE +
                       ((((size_t)w * NHEAD + hh) * NTOK + t) * DHEAD + dd);
          outH[idx] = (_Float16)val;
        } else if constexpr (EPI == 1) {      // PROJ: window-reverse + roll + residual
          int w = m / NTOK, t = m % NTOK;
          int bb = w >> 6, wi = w & 63;
          int wy = wi >> 3, wx = wi & 7, iy = t / WSZ, ix = t % WSZ;
          int y  = wy * WSZ + iy + SSZ; if (y  >= HH) y  -= HH;   // roll(+3)
          int xp = wx * WSZ + ix + SSZ; if (xp >= HH) xp -= HH;
          size_t idx = ((size_t)bb * (HH * HH) + (size_t)y * HH + xp) * CC + n;
          outF[idx] = val + resid[idx];
        } else if constexpr (EPI == 2) {      // FC1 + exact GELU (row stride == NTOT)
          float gl = 0.5f * val * (1.0f + erff(val * 0.70710678118654752f));
          outH[(size_t)m * NTOT + n] = (_Float16)gl;
        } else {                              // FC2 + residual accumulate
          outF[(size_t)m * CC + n] += val;
        }
      }
}

// ---------------------------------------------------------------------------
// Attention: one (window, head) per 128-thread block.
// q/k staged via async global->LDS (contiguous 49x32 f16 = 3136B), v scalar
// transposed. S = q@k^T via WMMA (49->64 pad), +bias +mask, softmax, O = P@V.
// ---------------------------------------------------------------------------
__global__ __launch_bounds__(128) void attn_kernel(const _Float16* __restrict__ qkv,
                                                   const float* __restrict__ rpb,
                                                   _Float16* __restrict__ outO) {
  __shared__ __attribute__((aligned(16))) _Float16 lq[64 * 32];
  __shared__ __attribute__((aligned(16))) _Float16 lk[64 * 32];
  __shared__ __attribute__((aligned(16))) _Float16 lvt[32 * 64];   // [d][m]
  __shared__ __attribute__((aligned(16))) float    lS[64 * 64];
  __shared__ __attribute__((aligned(16))) _Float16 lP[64 * 64];

  const int wh = blockIdx.x;
  const int w = wh >> 3, h = wh & 7;
  const int tid = threadIdx.x;
  const _Float16* qp = qkv + ((size_t)w * NHEAD + h) * NTOK * DHEAD;
  const _Float16* kp = qp + QKV_STRIDE;
  const _Float16* vp = qp + 2 * QKV_STRIDE;

  // q/k: 1568 valid halves each = 196 x 16B chunks, async DMA into LDS.
  for (int e = tid; e < 196; e += 128) {
    async_b128(lds_addr32(lq) + e * 16, qp + e * 8);
    async_b128(lds_addr32(lk) + e * 16, kp + e * 8);
  }
  // zero-pad rows 49..63 of q/k
  for (int e = 1568 + tid; e < 2048; e += 128) {
    lq[e] = (_Float16)0.0f; lk[e] = (_Float16)0.0f;
  }
  // v: transpose into [d][m], zero-padded
  for (int e = tid; e < 2048; e += 128) {
    int r = e >> 5, dd = e & 31;
    lvt[dd * 64 + r] = (r < NTOK) ? vp[e] : (_Float16)0.0f;
  }
  wait_async0();
  __syncthreads();

  const int wave = tid >> 5, lane = tid & 31;
  const int r0 = wave * 16;
  const int lrow = lane & 15, lhi = lane >> 4;

  // ---- S = q @ k^T ----
  Frag16 aq;
  aq.h[0] = *(const v8h*)(lq + (r0 + lrow) * 32 + lhi * 8);
  aq.h[1] = *(const v8h*)(lq + (r0 + lrow) * 32 + 16 + lhi * 8);
#pragma unroll
  for (int nt = 0; nt < 4; ++nt) {
    Frag16 bk;
    bk.h[0] = *(const v8h*)(lk + (nt * 16 + lrow) * 32 + lhi * 8);
    bk.h[1] = *(const v8h*)(lk + (nt * 16 + lrow) * 32 + 16 + lhi * 8);
    v8f s = {};
    s = wmma_f16(aq, bk, s);
#pragma unroll
    for (int r = 0; r < 8; ++r)
      lS[(r0 + r + lhi * 8) * 64 + nt * 16 + lrow] = s[r];
  }
  __syncthreads();

  // ---- bias + mask + softmax (wave handles its 16 rows; 2 cols/lane) ----
  const int wi = w & 63, wy = wi >> 3, wx = wi & 7;
  for (int j = 0; j < 16; ++j) {
    int rr = r0 + j;
    int tr = rr < NTOK ? rr : NTOK - 1;   // clamp padded rows for table index
    int riy = tr / WSZ, rix = tr % WSZ;
    int ry = wy * WSZ + riy, rx = wx * WSZ + rix;
    int regr = (ry < 49 ? 0 : (ry < 52 ? 1 : 2)) * 3 + (rx < 49 ? 0 : (rx < 52 ? 1 : 2));
    float vals[2];
#pragma unroll
    for (int s = 0; s < 2; ++s) {
      int c = lane + s * 32;
      float vv = -1e30f;
      if (c < NTOK) {
        int ciy = c / WSZ, cix = c % WSZ;
        int rel = (riy - ciy + WSZ - 1) * (2 * WSZ - 1) + (rix - cix + WSZ - 1);
        float bval = rpb[rel * NHEAD + h];
        int cy = wy * WSZ + ciy, cx = wx * WSZ + cix;
        int regc = (cy < 49 ? 0 : (cy < 52 ? 1 : 2)) * 3 + (cx < 49 ? 0 : (cx < 52 ? 1 : 2));
        float msk = (regc == regr) ? 0.0f : -100.0f;
        vv = lS[rr * 64 + c] + bval + msk;
      }
      vals[s] = vv;
    }
    float mx = fmaxf(vals[0], vals[1]);
    for (int off = 16; off; off >>= 1) mx = fmaxf(mx, __shfl_xor(mx, off));
    float e0 = (vals[0] > -1e29f) ? __expf(vals[0] - mx) : 0.0f;
    float e1 = (vals[1] > -1e29f) ? __expf(vals[1] - mx) : 0.0f;
    float sum = e0 + e1;
    for (int off = 16; off; off >>= 1) sum += __shfl_xor(sum, off);
    float inv = 1.0f / sum;
    lP[rr * 64 + lane]      = (_Float16)(e0 * inv);
    lP[rr * 64 + lane + 32] = (_Float16)(e1 * inv);
  }
  __syncthreads();

  // ---- O = P @ V  (K = 64 in two WMMA steps) ----
  v8f oacc[2] = {};
#pragma unroll
  for (int ks = 0; ks < 2; ++ks) {
    Frag16 pa;
    pa.h[0] = *(const v8h*)(lP + (r0 + lrow) * 64 + ks * 32 + lhi * 8);
    pa.h[1] = *(const v8h*)(lP + (r0 + lrow) * 64 + ks * 32 + 16 + lhi * 8);
#pragma unroll
    for (int nt = 0; nt < 2; ++nt) {
      Frag16 vb;
      vb.h[0] = *(const v8h*)(lvt + (nt * 16 + lrow) * 64 + ks * 32 + lhi * 8);
      vb.h[1] = *(const v8h*)(lvt + (nt * 16 + lrow) * 64 + ks * 32 + 16 + lhi * 8);
      oacc[nt] = wmma_f16(pa, vb, oacc[nt]);
    }
  }
#pragma unroll
  for (int nt = 0; nt < 2; ++nt)
#pragma unroll
    for (int r = 0; r < 8; ++r) {
      int rr = r0 + r + lhi * 8;
      if (rr < NTOK) {
        int n = nt * 16 + lrow;
        outO[((size_t)w * NTOK + rr) * CC + h * DHEAD + n] = (_Float16)oacc[nt][r];
      }
    }
}

// ---------------------------------------------------------------------------
// Launch sequence. Workspace layout (bytes, aliased):
//   [0, 2MB)        f16 weights
//   [2MB, 100MB)   LN1/LN2 activations (f16)
//   [100MB, ~492MB) q/k/v + attention-out; FC1 output aliases this region.
// ---------------------------------------------------------------------------
extern "C" void kernel_launch(void* const* d_in, const int* in_sizes, int n_in,
                              void* d_out, int out_size, void* d_ws, size_t ws_size,
                              hipStream_t stream) {
  const float* x     = (const float*)d_in[0];
  const float* n1g   = (const float*)d_in[1];
  const float* n1b   = (const float*)d_in[2];
  const float* qkvw  = (const float*)d_in[3];
  const float* qkvb  = (const float*)d_in[4];
  const float* rpb   = (const float*)d_in[5];
  const float* projw = (const float*)d_in[6];
  const float* projb = (const float*)d_in[7];
  const float* n2g   = (const float*)d_in[8];
  const float* n2b   = (const float*)d_in[9];
  const float* fc1w  = (const float*)d_in[10];
  const float* fc1b  = (const float*)d_in[11];
  const float* fc2w  = (const float*)d_in[12];
  const float* fc2b  = (const float*)d_in[13];
  float* out = (float*)d_out;
  char* ws = (char*)d_ws;

  _Float16* wQKV  = (_Float16*)(ws);
  _Float16* wPROJ = (_Float16*)(ws + 393216);
  _Float16* wFC1  = (_Float16*)(ws + 524288);
  _Float16* wFC2  = (_Float16*)(ws + 1048576);
  _Float16* Aact  = (_Float16*)(ws + 2097152ull);
  _Float16* qkvB  = (_Float16*)(ws + 104857600ull);
  _Float16* Oact  = (_Float16*)(ws + 413138944ull);
  _Float16* H1    = (_Float16*)(ws + 104857600ull);  // aliases qkv+O (dead by then)

  // 1) weights -> f16
  f32_to_f16_kernel<<<(196608 + 255) / 256, 256, 0, stream>>>(qkvw, wQKV, 196608);
  f32_to_f16_kernel<<<(65536  + 255) / 256, 256, 0, stream>>>(projw, wPROJ, 65536);
  f32_to_f16_kernel<<<(262144 + 255) / 256, 256, 0, stream>>>(fc1w, wFC1, 262144);
  f32_to_f16_kernel<<<(262144 + 255) / 256, 256, 0, stream>>>(fc2w, wFC2, 262144);

  // 2) LN1 + shift + window partition
  ln_kernel<true><<<MTOT, 256, 0, stream>>>(x, n1g, n1b, Aact);

  // 3) QKV GEMM (200704x256 @ 256x768)
  gemm_wmma_kernel<0, 256, 768><<<dim3(12, MTOT / 128), 256, 0, stream>>>(
      Aact, wQKV, qkvb, nullptr, qkvB, nullptr);

  // 4) attention: 4096 windows * 8 heads
  attn_kernel<<<4096 * NHEAD, 128, 0, stream>>>(qkvB, rpb, Oact);

  // 5) proj GEMM + window reverse + roll + residual -> d_out
  gemm_wmma_kernel<1, 256, 256><<<dim3(4, MTOT / 128), 256, 0, stream>>>(
      Oact, wPROJ, projb, out, nullptr, x);

  // 6) LN2 over d_out
  ln_kernel<false><<<MTOT, 256, 0, stream>>>(out, n2g, n2b, Aact);

  // 7) FC1 + GELU (200704x256 @ 256x1024)
  gemm_wmma_kernel<2, 256, 1024><<<dim3(16, MTOT / 128), 256, 0, stream>>>(
      Aact, wFC1, fc1b, nullptr, H1, nullptr);

  // 8) FC2 + residual accumulate into d_out (200704x1024 @ 1024x256)
  gemm_wmma_kernel<3, 1024, 256><<<dim3(4, MTOT / 128), 256, 0, stream>>>(
      H1, wFC2, fc2b, out, nullptr, nullptr);
}
